// Model_63556926046584
// MI455X (gfx1250) — matrix operand order repaired and checked by reading, not compiled, not run on hardware
//
#include <hip/hip_runtime.h>
#include <stdint.h>

#define B_     2
#define LBLK   32
#define BLOCKN 64
#define MTOK   65
#define SEQ    (LBLK*MTOK)   /* 2080 */
#define DIM_   512
#define HEADS_ 8
#define HD_    64
#define FFN_   1024
#define TOK    (B_*SEQ)      /* 4160 */
#define NL     2

typedef __bf16 bf16;
typedef __attribute__((ext_vector_type(16))) __bf16 v16bf;
typedef __attribute__((ext_vector_type(8)))  __bf16 v8bf;
typedef __attribute__((ext_vector_type(8)))  float  v8f;

union V16U { v16bf v; v8bf h2[2]; unsigned short s[16]; };

__device__ __forceinline__ unsigned short f2bf(float f){
  unsigned u = __float_as_uint(f);
  unsigned r = u + 0x7FFFu + ((u >> 16) & 1u);
  return (unsigned short)(r >> 16);
}

__device__ __forceinline__ v8bf ld8bf(const unsigned short* p){
  v8bf r; __builtin_memcpy(&r, p, 16); return r;
}

// A-matrix bf16 16x32 fragment (ISA 7.12.2): lanes0-15 K{0..7,16..23}, lanes16-31 K{8..15,24..31}
__device__ __forceinline__ v16bf ldA(const unsigned short* row, int k0, int half){
  V16U a;
  a.h2[0] = ld8bf(row + k0 + half*8);
  a.h2[1] = ld8bf(row + k0 + half*8 + 16);
  return a.v;
}
// B-matrix bf16 32x16 fragment: lane=N, lanes0-15 K{0..7,16..23}, lanes16-31 K{8..15,24..31}
__device__ __forceinline__ v16bf ldB(const unsigned short* row, int k0, int half){
  V16U b;
  b.h2[0] = ld8bf(row + k0 + half*8);
  b.h2[1] = ld8bf(row + k0 + half*8 + 16);
  return b.v;
}

// ---------------- h = [blocks of x ; router token] ----------------
__global__ void k_build_h(const float* __restrict__ x, const float* __restrict__ rt,
                          float* __restrict__ h){
  int i = blockIdx.x*256 + threadIdx.x;
  if (i >= TOK*DIM_) return;
  int d = i % DIM_; int t = (i / DIM_) % SEQ; int b = i / (DIM_*SEQ);
  int l = t / MTOK, r = t % MTOK;
  h[i] = (r == BLOCKN) ? rt[d] : x[((size_t)(b*LBLK + l)*BLOCKN + r)*DIM_ + d];
}

// ---------------- fp32 [K,N] -> bf16 transposed [N,K] ----------------
__global__ void k_wconv(const float* __restrict__ W, unsigned short* __restrict__ Wt,
                        int K, int N){
  int i = blockIdx.x*256 + threadIdx.x;
  if (i >= K*N) return;
  int r = i / N, c = i % N;
  Wt[(size_t)c*K + r] = f2bf(W[i]);
}

// ---------------- rms_norm -> bf16 ----------------
__global__ void k_rmsnorm(const float* __restrict__ h, const float* __restrict__ w,
                          unsigned short* __restrict__ hn){
  int tok = blockIdx.x, tid = threadIdx.x;
  const float* row = h + (size_t)tok*DIM_;
  float a0 = row[tid], a1 = row[tid+256];
  __shared__ float red[256];
  red[tid] = a0*a0 + a1*a1;
  __syncthreads();
  for (int st = 128; st > 0; st >>= 1){
    if (tid < st) red[tid] += red[tid+st];
    __syncthreads();
  }
  float r = rsqrtf(red[0]/(float)DIM_ + 1e-5f);
  hn[(size_t)tok*DIM_ + tid]       = f2bf(a0*r*w[tid]);
  hn[(size_t)tok*DIM_ + tid + 256] = f2bf(a1*r*w[tid+256]);
}

// ---------------- WMMA GEMM, 32x32 macro-tile per wave (2x2 register blocking) -------
// C[T,N] = A[T,K](bf16) @ Wt[N,K](bf16)^T (+ optional fp32 residual)
__global__ void __launch_bounds__(256)
k_gemm(const unsigned short* __restrict__ A, const unsigned short* __restrict__ Wt,
       float* __restrict__ C, const float* __restrict__ Crd,
       int K, int N, int tilesN2, int totalTiles2){
  int wave = (int)((blockIdx.x*blockDim.x + threadIdx.x) >> 5);
  if (wave >= totalTiles2) return;
  int tm = wave / tilesN2, tn = wave % tilesN2;
  int lane = threadIdx.x & 31, half = lane >> 4, lm = lane & 15;
  const unsigned short* a0p = A  + (size_t)(tm*32 + lm)*K;
  const unsigned short* a1p = A  + (size_t)(tm*32 + 16 + lm)*K;
  const unsigned short* b0p = Wt + (size_t)(tn*32 + lm)*K;
  const unsigned short* b1p = Wt + (size_t)(tn*32 + 16 + lm)*K;
  int col0 = tn*32 + lm, col1 = col0 + 16;
  v8f acc00, acc01, acc10, acc11;
  if (Crd){
    for (int e = 0; e < 8; e++){
      size_t r0 = (size_t)(tm*32 + e + half*8)*N;
      size_t r1 = (size_t)(tm*32 + 16 + e + half*8)*N;
      acc00[e] = Crd[r0 + col0]; acc01[e] = Crd[r0 + col1];
      acc10[e] = Crd[r1 + col0]; acc11[e] = Crd[r1 + col1];
    }
  } else {
    for (int e = 0; e < 8; e++){
      acc00[e] = 0.f; acc01[e] = 0.f; acc10[e] = 0.f; acc11[e] = 0.f;
    }
  }
  for (int k0 = 0; k0 < K; k0 += 32){
    v16bf a0 = ldA(a0p, k0, half);
    v16bf a1 = ldA(a1p, k0, half);
    v16bf b0 = ldB(b0p, k0, half);
    v16bf b1 = ldB(b1p, k0, half);
    acc00 = __builtin_amdgcn_wmma_f32_16x16x32_bf16(false, a0, false, b0, (short)0, acc00, false, false);
    acc01 = __builtin_amdgcn_wmma_f32_16x16x32_bf16(false, a0, false, b1, (short)0, acc01, false, false);
    acc10 = __builtin_amdgcn_wmma_f32_16x16x32_bf16(false, a1, false, b0, (short)0, acc10, false, false);
    acc11 = __builtin_amdgcn_wmma_f32_16x16x32_bf16(false, a1, false, b1, (short)0, acc11, false, false);
  }
  for (int e = 0; e < 8; e++){
    size_t r0 = (size_t)(tm*32 + e + half*8)*N;
    size_t r1 = (size_t)(tm*32 + 16 + e + half*8)*N;
    C[r0 + col0] = acc00[e]; C[r0 + col1] = acc01[e];
    C[r1 + col0] = acc10[e]; C[r1 + col1] = acc11[e];
  }
}

// ---------------- qkv fp32 [T,1536] -> rotary(q,k) bf16 [B,H,S,64], vT bf16 [B,H,64,S] --
__global__ void k_qkvpost(const float* __restrict__ qkv,
                          unsigned short* __restrict__ qb,
                          unsigned short* __restrict__ kb,
                          unsigned short* __restrict__ vT){
  int tok = blockIdx.x;                 // 0..TOK-1
  int b = tok / SEQ, s = tok % SEQ;
  int h = threadIdx.x >> 5, j = threadIdx.x & 31;
  const float* base = qkv + (size_t)tok*(3*DIM_);
  float inv = __powf(10000.f, -(float)j/32.f);
  float ang = (float)s * inv;
  float c = __cosf(ang), sn = __sinf(ang);
  size_t qoff = ((size_t)(b*HEADS_ + h)*SEQ + s)*HD_;
  float q1 = base[h*HD_ + j], q2 = base[h*HD_ + j + 32];
  qb[qoff + j]      = f2bf( q1*c + q2*sn);
  qb[qoff + j + 32] = f2bf(-q1*sn + q2*c);
  float k1 = base[DIM_ + h*HD_ + j], k2 = base[DIM_ + h*HD_ + j + 32];
  kb[qoff + j]      = f2bf( k1*c + k2*sn);
  kb[qoff + j + 32] = f2bf(-k1*sn + k2*c);
  size_t vbase = (size_t)(b*HEADS_ + h)*HD_*SEQ;
  vT[vbase + (size_t)j*SEQ + s]        = f2bf(base[2*DIM_ + h*HD_ + j]);
  vT[vbase + (size_t)(j+32)*SEQ + s]   = f2bf(base[2*DIM_ + h*HD_ + j + 32]);
}

// ---------------- flash attention (WMMA QK^T and PV), mask = block-causal & same-doc --
__global__ void __launch_bounds__(256)
k_flash(const unsigned short* __restrict__ qb, const unsigned short* __restrict__ kb,
        const unsigned short* __restrict__ vT, const int* __restrict__ doc,
        float* __restrict__ o){
  __shared__ unsigned short pls[8][16*32];
  int w  = threadIdx.x >> 5;
  int gw = blockIdx.x*8 + w;
  const int TQ = SEQ/16;                 // 130
  if (gw >= B_*HEADS_*TQ) return;
  int qt = gw % TQ;
  int h  = (gw / TQ) % HEADS_;
  int b  = gw / (TQ*HEADS_);
  int lane = threadIdx.x & 31, half = lane >> 4, lm = lane & 15;
  int qr0 = qt*16;

  const unsigned short* qrow = qb + ((size_t)(b*HEADS_ + h)*SEQ + qr0 + lm)*HD_;
  v16bf qa0 = ldA(qrow, 0, half);
  v16bf qa1 = ldA(qrow, 32, half);
  int dq[8], qblk[8];
  for (int e = 0; e < 8; e++){
    int qg = qr0 + e + half*8;
    dq[e]   = doc[b*SEQ + qg];
    qblk[e] = qg / MTOK;
  }
  int kend = ((qr0 + 15)/MTOK + 1)*MTOK;       // block-causal extent (<= SEQ)
  int nkt  = (kend + 15)/16;

  float mrow[8], lrow[8];
  for (int e = 0; e < 8; e++){ mrow[e] = -1e30f; lrow[e] = 0.f; }
  v8f oacc[4];
  for (int c = 0; c < 4; c++) for (int e = 0; e < 8; e++) oacc[c][e] = 0.f;
  const float scale = 0.125f;                  // 1/sqrt(64)

  auto compute_scores = [&](int kt) -> v8f {
    v8f s;
    if (kt < nkt){
      for (int e = 0; e < 8; e++) s[e] = 0.f;
      const unsigned short* krow = kb + ((size_t)(b*HEADS_ + h)*SEQ + kt*16 + lm)*HD_;
      s = __builtin_amdgcn_wmma_f32_16x16x32_bf16(false, qa0, false, ldB(krow, 0, half),
                                                  (short)0, s, false, false);
      s = __builtin_amdgcn_wmma_f32_16x16x32_bf16(false, qa1, false, ldB(krow, 32, half),
                                                  (short)0, s, false, false);
      int kg   = kt*16 + lm;
      int dk   = doc[b*SEQ + kg];
      int kblk = kg / MTOK;
      for (int e = 0; e < 8; e++){
        float sv = s[e]*scale;
        s[e] = ((qblk[e] >= kblk) && (dq[e] == dk)) ? sv : -1e30f;
      }
    } else {
      for (int e = 0; e < 8; e++) s[e] = -1e30f;
    }
    return s;
  };

  for (int kt2 = 0; kt2 < nkt; kt2 += 2){
    v8f s0 = compute_scores(kt2);
    v8f s1 = compute_scores(kt2 + 1);
    for (int e = 0; e < 8; e++){
      float mt = fmaxf(s0[e], s1[e]);
      for (int d2 = 1; d2 < 16; d2 <<= 1) mt = fmaxf(mt, __shfl_xor(mt, d2, 16));
      float mnew = fmaxf(mrow[e], mt);
      float corr = __expf(mrow[e] - mnew);
      float p0 = __expf(s0[e] - mnew);
      float p1 = __expf(s1[e] - mnew);
      float rs = p0 + p1;
      for (int d2 = 1; d2 < 16; d2 <<= 1) rs += __shfl_xor(rs, d2, 16);
      lrow[e] = lrow[e]*corr + rs;
      mrow[e] = mnew;
      for (int c = 0; c < 4; c++) oacc[c][e] *= corr;
      int r = (e + half*8)*32;
      pls[w][r + lm]       = f2bf(p0);
      pls[w][r + 16 + lm]  = f2bf(p1);
    }
    // P (C-layout in LDS) -> A-layout fragment, then PV WMMA over 32 keys
    V16U pa;
    for (int e = 0; e < 16; e++){
      int kidx = e + half*8 + ((e >= 8) ? 8 : 0);
      pa.s[e] = pls[w][lm*32 + kidx];
    }
    for (int c = 0; c < 4; c++){
      const unsigned short* vrow =
          vT + ((size_t)(b*HEADS_ + h)*HD_ + c*16 + lm)*SEQ + kt2*16;
      oacc[c] = __builtin_amdgcn_wmma_f32_16x16x32_bf16(false, pa.v, false, ldB(vrow, 0, half),
                                                        (short)0, oacc[c], false, false);
    }
  }
  for (int e = 0; e < 8; e++){
    int qg = qr0 + e + half*8;
    float invl = 1.f / lrow[e];
    for (int c = 0; c < 4; c++)
      o[((size_t)(b*SEQ + qg))*DIM_ + h*HD_ + c*16 + lm] = oacc[c][e]*invl;
  }
}

// ---------------- misc elementwise ----------------
__global__ void k_f2bf(const float* __restrict__ src, unsigned short* __restrict__ dst, int n){
  int i = blockIdx.x*256 + threadIdx.x;
  if (i < n) dst[i] = f2bf(src[i]);
}

__global__ void k_silu(const float* __restrict__ u, unsigned short* __restrict__ g){
  int i = blockIdx.x*256 + threadIdx.x;
  if (i >= TOK*FFN_) return;
  int t = i / FFN_, j = i % FFN_;
  float a  = u[(size_t)t*2*FFN_ + j];
  float bb = u[(size_t)t*2*FFN_ + FFN_ + j];
  float sl = a / (1.f + __expf(-a));
  g[i] = f2bf(sl*bb);
}

__global__ void k_out(const float* __restrict__ h, float* __restrict__ out){
  int i = blockIdx.x*256 + threadIdx.x;
  if (i >= B_*LBLK*DIM_) return;
  int d = i % DIM_; int l = (i/DIM_) % LBLK; int b = i/(DIM_*LBLK);
  out[i] = h[((size_t)b*SEQ + l*MTOK + BLOCKN)*DIM_ + d];
}

// ---------------- orchestration ----------------
extern "C" void kernel_launch(void* const* d_in, const int* in_sizes, int n_in,
                              void* d_out, int out_size, void* d_ws, size_t ws_size,
                              hipStream_t stream) {
  const float* x          = (const float*)d_in[0];
  const int*   doc        = (const int*)  d_in[1];
  const float* rt         = (const float*)d_in[2];
  const float* attn_w     = (const float*)d_in[3];
  const float* attn_o_w   = (const float*)d_in[4];
  const float* ffn_up_w   = (const float*)d_in[5];
  const float* ffn_down_w = (const float*)d_in[6];
  const float* attn_nw    = (const float*)d_in[7];
  const float* ffn_nw     = (const float*)d_in[8];
  (void)in_sizes; (void)n_in; (void)out_size; (void)ws_size;

  char* ws = (char*)d_ws;
  size_t off = 0;
  auto alloc = [&](size_t bytes) -> void* {
    void* p = ws + off;
    off = (off + bytes + 255) & ~(size_t)255;
    return p;
  };
  float*          h    = (float*)         alloc((size_t)TOK*DIM_*4);
  unsigned short* hn   = (unsigned short*)alloc((size_t)TOK*DIM_*2);
  unsigned short* qbf  = (unsigned short*)alloc((size_t)B_*HEADS_*SEQ*HD_*2);
  unsigned short* kbf  = (unsigned short*)alloc((size_t)B_*HEADS_*SEQ*HD_*2);
  unsigned short* vT   = (unsigned short*)alloc((size_t)B_*HEADS_*HD_*SEQ*2);
  float*          obuf = (float*)         alloc((size_t)TOK*DIM_*4);
  unsigned short* obf  = (unsigned short*)alloc((size_t)TOK*DIM_*2);
  unsigned short* g    = (unsigned short*)alloc((size_t)TOK*FFN_*2);
  float*          big  = (float*)         alloc((size_t)TOK*2*FFN_*4);  // qkv & u (unioned)
  unsigned short* wt_attn = (unsigned short*)alloc((size_t)NL*1536*512*2);
  unsigned short* wt_o    = (unsigned short*)alloc((size_t)NL*512*512*2);
  unsigned short* wt_up   = (unsigned short*)alloc((size_t)NL*2048*512*2);
  unsigned short* wt_down = (unsigned short*)alloc((size_t)NL*512*1024*2);

  // weight conversion (fp32 [K,N] -> bf16 [N,K])
  for (int l = 0; l < NL; l++){
    k_wconv<<<(512*1536+255)/256, 256, 0, stream>>>(attn_w     + (size_t)l*512*1536,
                                                    wt_attn    + (size_t)l*1536*512, 512, 1536);
    k_wconv<<<(512*512 +255)/256, 256, 0, stream>>>(attn_o_w   + (size_t)l*512*512,
                                                    wt_o       + (size_t)l*512*512,  512, 512);
    k_wconv<<<(512*2048+255)/256, 256, 0, stream>>>(ffn_up_w   + (size_t)l*512*2048,
                                                    wt_up      + (size_t)l*2048*512, 512, 2048);
    k_wconv<<<(1024*512+255)/256, 256, 0, stream>>>(ffn_down_w + (size_t)l*1024*512,
                                                    wt_down    + (size_t)l*512*1024, 1024, 512);
  }

  k_build_h<<<(TOK*DIM_+255)/256, 256, 0, stream>>>(x, rt, h);

  const int TM2 = TOK/32;   // 130 macro-tiles of 32 rows
  for (int l = 0; l < NL; l++){
    // attention block
    k_rmsnorm<<<TOK, 256, 0, stream>>>(h, attn_nw + l*DIM_, hn);
    {
      int tilesN2 = 1536/32, tiles = TM2*tilesN2;
      k_gemm<<<(tiles+7)/8, 256, 0, stream>>>(hn, wt_attn + (size_t)l*1536*512,
                                              big, nullptr, 512, 1536, tilesN2, tiles);
    }
    k_qkvpost<<<TOK, 256, 0, stream>>>(big, qbf, kbf, vT);
    k_flash<<<(B_*HEADS_*(SEQ/16)+7)/8, 256, 0, stream>>>(qbf, kbf, vT, doc, obuf);
    k_f2bf<<<(TOK*DIM_+255)/256, 256, 0, stream>>>(obuf, obf, TOK*DIM_);
    {
      int tilesN2 = 512/32, tiles = TM2*tilesN2;
      k_gemm<<<(tiles+7)/8, 256, 0, stream>>>(obf, wt_o + (size_t)l*512*512,
                                              h, h, 512, 512, tilesN2, tiles);
    }
    // ffn block
    k_rmsnorm<<<TOK, 256, 0, stream>>>(h, ffn_nw + l*DIM_, hn);
    {
      int tilesN2 = 2048/32, tiles = TM2*tilesN2;
      k_gemm<<<(tiles+7)/8, 256, 0, stream>>>(hn, wt_up + (size_t)l*2048*512,
                                              big, nullptr, 512, 2048, tilesN2, tiles);
    }
    k_silu<<<(TOK*FFN_+255)/256, 256, 0, stream>>>(big, g);
    {
      int tilesN2 = 512/32, tiles = TM2*tilesN2;
      k_gemm<<<(tiles+7)/8, 256, 0, stream>>>(g, wt_down + (size_t)l*512*1024,
                                              h, h, 1024, 512, tilesN2, tiles);
    }
  }
  k_out<<<(B_*LBLK*DIM_+255)/256, 256, 0, stream>>>(h, (float*)d_out);
}
